// SparseLookupFFNv2_51934744543475
// MI455X (gfx1250) — matrix operand order repaired and checked by reading, not compiled, not run on hardware
//
#include <hip/hip_runtime.h>
#include <hip/hip_bf16.h>

// ---------------------------------------------------------------------------
// SparseLookupFFNv2 fused kernel for gfx1250 (MI455X), wave32 + WMMA bf16.
//
// Roofline: ~200 MB HBM traffic (x in, y out, x re-read) -> ~9us floor at
// 23.3 TB/s. All three GEMMs (tile scores, cluster scores, compress-1) fold
// into one x-tile GEMM against a combined B matrix [336 x 1024] bf16
// (tile sigs | cluster sigs | pad | W1^T). 32 tokens per block: each B
// fragment loaded from L2 feeds TWO v_wmma_f32_16x16x32_bf16 (two M-tiles in
// registers), halving L2 B traffic vs 16-token blocks. ~110 KB dynamic LDS
// per block (gfx1250 WGP has 320 KB). Calibration spline is monotonic
// increasing -> argmax(gates) == argmax(raw scores), so knots/temperature
// drop out of routing entirely.
// ---------------------------------------------------------------------------

typedef __attribute__((ext_vector_type(8)))  __bf16 v8bf;
typedef __attribute__((ext_vector_type(16))) __bf16 v16bf;
typedef __attribute__((ext_vector_type(8)))  float  v8f;

#define D_MODEL    1024
#define NUM_TILES  64
#define TPC        8
#define GRID_G     16
#define COMP_H     256
#define SCORE_COLS 80                      // 64 tile + 8 cluster + 8 zero pad
#define NCOLS      (SCORE_COLS + COMP_H)   // 336
#define NTILES_N   (NCOLS / 16)            // 21
#define ROWS       32                      // tokens per block (two WMMA M-tiles)
#define XSTRIDE    (D_MODEL + 8)           // bf16 halfs, padded row stride
#define OSTRIDE    340                     // f32 cols, padded (336 used)
#define XBF_BYTES  (ROWS * XSTRIDE * 2)            // 66,048
#define SMEM_BYTES (XBF_BYTES + ROWS * OSTRIDE * 4) // 109,568

__device__ __forceinline__ float ternq(float v) {
    return v > 0.3f ? 1.0f : (v < -0.3f ? -1.0f : 0.0f);
}

__device__ __forceinline__ float wred(float v) {
    #pragma unroll
    for (int off = 16; off > 0; off >>= 1) v += __shfl_xor(v, off, 32);
    return v;
}

// ---------------------------------------------------------------------------
// Build combined B^T [NCOLS][D_MODEL] in bf16:
//   rows [0,64)   : ternary-quantized signatures
//   rows [64,72)  : cluster signatures = sign(sum of member ternary sigs)
//   rows [72,80)  : zero padding (score block = 5 clean 16-col WMMA tiles)
//   rows [80,336) : compress W1 transposed (W1 is [D_MODEL][COMP_H])
// ---------------------------------------------------------------------------
__global__ void prep_B(const float* __restrict__ sigraw,
                       const float* __restrict__ cw1,
                       __bf16* __restrict__ Bt) {
    int e = blockIdx.x * 256 + threadIdx.x;
    if (e >= NCOLS * D_MODEL) return;
    int n = e >> 10;
    int k = e & (D_MODEL - 1);
    float v;
    if (n < NUM_TILES) {
        v = ternq(sigraw[n * D_MODEL + k]);
    } else if (n < NUM_TILES + 8) {
        int c = n - NUM_TILES;
        float s = 0.0f;
        #pragma unroll
        for (int j = 0; j < TPC; ++j) s += ternq(sigraw[(c * TPC + j) * D_MODEL + k]);
        v = s > 0.0f ? 1.0f : (s < 0.0f ? -1.0f : 0.0f);   // sign(mean)==sign(sum)
    } else if (n < SCORE_COLS) {
        v = 0.0f;
    } else {
        v = cw1[k * COMP_H + (n - SCORE_COLS)];            // W1^T
    }
    Bt[e] = (__bf16)v;
}

// ---------------------------------------------------------------------------
// Fused main kernel: 32 tokens per block, 256 threads (8 wave32s).
// ---------------------------------------------------------------------------
__global__ void __launch_bounds__(256)
fused_ffn(const float* __restrict__ xg,
          const __bf16* __restrict__ Bt,
          const float* __restrict__ cb1,
          const float* __restrict__ cw2,
          const float* __restrict__ cb2,
          const float* __restrict__ sw1,
          const float* __restrict__ sb1,
          const float* __restrict__ sw2,
          const float* __restrict__ sb2,
          const float* __restrict__ dirs,
          const float* __restrict__ gamma,
          const float* __restrict__ beta,
          const float* __restrict__ oscale_p,
          float* __restrict__ outp,
          int N) {
    extern __shared__ char smem[];
    __bf16* xbf  = (__bf16*)smem;                  // [ROWS][XSTRIDE] bf16
    float*  out_s = (float*)(smem + XBF_BYTES);    // [ROWS][OSTRIDE] f32

    const int tid  = threadIdx.x;
    const int wid  = tid >> 5;
    const int lane = tid & 31;
    const int tok_base = blockIdx.x * ROWS;

    // ---- Stage 0: load + convert x tile f32 -> bf16 into LDS -------------
    // 32 rows x 256 float4 = 8192 float4, 32 per thread.
    for (int q = tid; q < ROWS * (D_MODEL / 4); q += 256) {
        int row = q >> 8;
        int c4  = q & 255;
        int tok = tok_base + row;
        float4 xv = make_float4(0.f, 0.f, 0.f, 0.f);
        if (tok < N) xv = ((const float4*)(xg + (size_t)tok * D_MODEL))[c4];
        __bf16* d = &xbf[row * XSTRIDE + c4 * 4];
        d[0] = (__bf16)xv.x; d[1] = (__bf16)xv.y;
        d[2] = (__bf16)xv.z; d[3] = (__bf16)xv.w;
    }
    __syncthreads();

    // ---- Stage 1: [32 x 1024] @ [1024 x 336] via WMMA bf16 ---------------
    // A layout (16-bit 16x32): lane<16 -> row=lane, K k..k+15;
    //                          lane>=16 -> row=lane-16, K k+16..k+31.
    // B layout (32x16): lane<16 -> col=lane, K k..k+15 ; lane>=16 -> K k+16..k+31.
    // Each B fragment (global/L2) feeds two WMMAs: M-tile 0 (tokens 0-15)
    // and M-tile 1 (tokens 16-31).
    {
        const int  lrow = lane & 15;
        const bool hi   = lane >= 16;
        const __bf16* arow0 = &xbf[lrow * XSTRIDE + (hi ? 16 : 0)];
        const __bf16* arow1 = &xbf[(16 + lrow) * XSTRIDE + (hi ? 16 : 0)];
        for (int nt = wid; nt < NTILES_N; nt += 8) {
            const __bf16* brow = Bt + (size_t)(nt * 16 + lrow) * D_MODEL + (hi ? 16 : 0);
            v8f c0 = {}, c1 = {};
            #pragma unroll 4
            for (int k = 0; k < D_MODEL; k += 32) {
                v16bf b = *(const v16bf*)(brow + k);
                v8bf a0 = *(const v8bf*)(arow0 + k);
                v8bf a1 = *(const v8bf*)(arow0 + k + 8);
                v16bf aA = __builtin_shufflevector(a0, a1,
                           0,1,2,3,4,5,6,7,8,9,10,11,12,13,14,15);
                c0 = __builtin_amdgcn_wmma_f32_16x16x32_bf16(
                         false, aA, false, b, (short)0, c0, false, false);
                v8bf a2 = *(const v8bf*)(arow1 + k);
                v8bf a3 = *(const v8bf*)(arow1 + k + 8);
                v16bf aB = __builtin_shufflevector(a2, a3,
                           0,1,2,3,4,5,6,7,8,9,10,11,12,13,14,15);
                c1 = __builtin_amdgcn_wmma_f32_16x16x32_bf16(
                         false, aB, false, b, (short)0, c1, false, false);
            }
            // D layout: VGPR r -> (row = r + (hi?8:0), col = lrow)
            const int row0 = hi ? 8 : 0;
            const int col  = nt * 16 + lrow;
            #pragma unroll
            for (int r = 0; r < 8; ++r) {
                out_s[(row0 + r) * OSTRIDE + col]        = c0[r];
                out_s[(16 + row0 + r) * OSTRIDE + col]   = c1[r];
            }
        }
    }
    __syncthreads();

    // ---- Stage 2: compress-2, routing, spline, residual + LayerNorm ------
    const float oscale = oscale_p[0];
    #pragma unroll
    for (int tt = 0; tt < 4; ++tt) {
        const int t   = wid * 4 + tt;           // local token in [0,32)
        const int tok = tok_base + t;
        if (tok >= N) continue;

        // compress layer 2: comp = tanh(gelu(h + b1) @ W2 + b2)
        float s0 = 0.0f, s1 = 0.0f;
        #pragma unroll
        for (int i = 0; i < COMP_H / 32; ++i) {
            int j = lane + 32 * i;
            float hv  = out_s[t * OSTRIDE + SCORE_COLS + j] + cb1[j];
            float act = 0.5f * hv * (1.0f + erff(hv * 0.70710678f));
            s0 = fmaf(act, cw2[2 * j + 0], s0);
            s1 = fmaf(act, cw2[2 * j + 1], s1);
        }
        s0 = wred(s0); s1 = wred(s1);
        const float comp0 = tanhf(s0 + cb2[0]);
        const float comp1 = tanhf(s1 + cb2[1]);

        // routing: calibration is monotonic -> argmax over raw scores.
        float best = -3.4e38f; int cbest = 0;
        #pragma unroll
        for (int c = 0; c < 8; ++c) {
            float v = out_s[t * OSTRIDE + NUM_TILES + c];
            if (v > best) { best = v; cbest = c; }
        }
        best = -3.4e38f; int jbest = 0;
        #pragma unroll
        for (int j = 0; j < TPC; ++j) {
            float v = out_s[t * OSTRIDE + cbest * TPC + j];
            if (v > best) { best = v; jbest = j; }
        }
        const int tile = cbest * TPC + jbest;

        // per-tile spline MLP: relu(comp @ w1 + b1) @ w2 + b2
        const int g = lane & 15;
        float act = fmaxf(0.0f, fmaf(comp0, sw1[tile * 32 + g],
                               fmaf(comp1, sw1[tile * 32 + GRID_G + g],
                                    sb1[tile * GRID_G + g])));
        float part = (lane < GRID_G) ? act * sw2[tile * GRID_G + g] : 0.0f;
        const float mag  = wred(part) + sb2[tile];
        const float coef = mag * oscale;

        // residual + LayerNorm over 1024 (32 lanes x 8 float4 each)
        const float4* xv = (const float4*)(xg   + (size_t)tok  * D_MODEL);
        const float4* dv = (const float4*)(dirs + (size_t)tile * D_MODEL);
        float4 f[8];
        float s = 0.0f, s2 = 0.0f;
        #pragma unroll
        for (int i = 0; i < 8; ++i) {
            int q = lane + 32 * i;
            float4 xi = xv[q], di = dv[q], fi;
            fi.x = fmaf(coef, di.x, xi.x);
            fi.y = fmaf(coef, di.y, xi.y);
            fi.z = fmaf(coef, di.z, xi.z);
            fi.w = fmaf(coef, di.w, xi.w);
            f[i] = fi;
            s  += fi.x + fi.y + fi.z + fi.w;
            s2 += fi.x * fi.x + fi.y * fi.y + fi.z * fi.z + fi.w * fi.w;
        }
        s = wred(s); s2 = wred(s2);
        const float mean = s * (1.0f / (float)D_MODEL);
        const float var  = s2 * (1.0f / (float)D_MODEL) - mean * mean;
        const float rstd = rsqrtf(var + 1e-5f);

        float4* ov = (float4*)(outp + (size_t)tok * D_MODEL);
        const float4* gv = (const float4*)gamma;
        const float4* bv = (const float4*)beta;
        #pragma unroll
        for (int i = 0; i < 8; ++i) {
            int q = lane + 32 * i;
            float4 g4 = gv[q], b4 = bv[q], fi = f[i], o;
            o.x = fmaf((fi.x - mean) * rstd, g4.x, b4.x);
            o.y = fmaf((fi.y - mean) * rstd, g4.y, b4.y);
            o.z = fmaf((fi.z - mean) * rstd, g4.z, b4.z);
            o.w = fmaf((fi.w - mean) * rstd, g4.w, b4.w);
            ov[q] = o;
        }
    }
}

extern "C" void kernel_launch(void* const* d_in, const int* in_sizes, int n_in,
                              void* d_out, int out_size, void* d_ws, size_t ws_size,
                              hipStream_t stream) {
    (void)n_in; (void)out_size; (void)ws_size;
    const float* xg     = (const float*)d_in[0];
    const float* sigraw = (const float*)d_in[1];
    // d_in[2] knot_values, d_in[3] temperature: unused (calibration is
    // monotonic, so argmax of calibrated gates == argmax of raw scores).
    const float* cw1    = (const float*)d_in[4];
    const float* cb1    = (const float*)d_in[5];
    const float* cw2    = (const float*)d_in[6];
    const float* cb2    = (const float*)d_in[7];
    const float* sw1    = (const float*)d_in[8];
    const float* sb1    = (const float*)d_in[9];
    const float* sw2    = (const float*)d_in[10];
    const float* sb2    = (const float*)d_in[11];
    const float* dirs   = (const float*)d_in[12];
    const float* gamma  = (const float*)d_in[13];
    const float* beta   = (const float*)d_in[14];
    const float* oscale = (const float*)d_in[15];
    float* out = (float*)d_out;

    const int N = in_sizes[0] / D_MODEL;          // 16384 tokens
    __bf16* Bt = (__bf16*)d_ws;                   // 336*1024*2 = 688,128 B

    // Allow >64KB dynamic LDS (gfx1250 WGP has 320KB). Idempotent host-side
    // call; harmless during graph capture.
    (void)hipFuncSetAttribute((const void*)fused_ffn,
                              hipFuncAttributeMaxDynamicSharedMemorySize,
                              SMEM_BYTES);

    const int prep_blocks = (NCOLS * D_MODEL + 255) / 256;
    prep_B<<<prep_blocks, 256, 0, stream>>>(sigraw, cw1, Bt);

    const int blocks = (N + ROWS - 1) / ROWS;     // 512
    fused_ffn<<<blocks, 256, SMEM_BYTES, stream>>>(xg, Bt, cb1, cw2, cb2,
                                                   sw1, sb1, sw2, sb2,
                                                   dirs, gamma, beta, oscale,
                                                   out, N);
}
